// ALIGNNFF2_60447369724153
// MI455X (gfx1250) — compile-verified, matrix operand order checked by reading, not run on hardware
//
#include <hip/hip_runtime.h>
#include <hip/hip_bf16.h>

// ---------------------------------------------------------------------------
// ALIGNN-FF graph block, algebraically collapsed:
//   segment_sum is linear and edge features are only used linearly, so the
//   two 600k-row edge GEMMs collapse into 50k-row node GEMMs after a single
//   scatter-add of e0 into S[dst]. Node tensors (25.6MB) live in the 192MB L2,
//   so HBM traffic ~= one read of e0 (307MB) -> ~14us floor at 23.3 TB/s.
//   GEMMs (~8 GFLOP) run on v_wmma_f32_16x16x32_bf16 (f32 accumulate).
//   This round: compile-time-specialized GEMM (no runtime flag branches) and
//   explicit b128 vector loads for both fragments.
// ---------------------------------------------------------------------------

typedef __attribute__((ext_vector_type(16))) __bf16 v16bf;
typedef __attribute__((ext_vector_type(8)))  __bf16 v8bf;
typedef __attribute__((ext_vector_type(8)))  float  v8f;
typedef __attribute__((ext_vector_type(4)))  float  v4f;

#define NNODES 50000
#define NEDGES 600000
#define NGRAPH 512
#define DD     128

// ---------------- weight prep: W[k][n] (f32) -> Wt[w][n][k] (bf16) ----------
__global__ void prep_weights(const float* __restrict__ W0, const float* __restrict__ W1,
                             const float* __restrict__ W2, const float* __restrict__ W3,
                             __bf16* __restrict__ Wt) {
    int t = blockIdx.x * blockDim.x + threadIdx.x;
    if (t >= 4 * DD * DD) return;
    int w = t >> 14;            // which weight matrix
    int rem = t & 16383;
    int n = rem >> 7;           // output column
    int k = rem & 127;          // k index
    const float* W = (w == 0) ? W0 : (w == 1) ? W1 : (w == 2) ? W2 : W3;
    Wt[(size_t)w * DD * DD + (size_t)n * DD + k] = (__bf16)W[(size_t)k * DD + n];
}

// ---------------- edge scatter: S[dst]+=E[e], G[dst]+=X[src], cnt[dst]+=1 ---
template <bool HAS_E, bool HAS_X, bool HAS_CNT>
__global__ void scatter_edges(const float* __restrict__ Efeat,
                              const float* __restrict__ Xfeat,
                              const int* __restrict__ src, const int* __restrict__ dst,
                              float* __restrict__ Sacc, float* __restrict__ Gacc,
                              float* __restrict__ cnt, int nEdges) {
    int e = blockIdx.x * (blockDim.x >> 5) + (threadIdx.x >> 5);
    if (e >= nEdges) return;
    int lane = threadIdx.x & 31;
    int d = dst[e];
    size_t dbase = (size_t)d * DD + lane * 4;
    if constexpr (HAS_E) {
        const v4f ev = *(const v4f*)(Efeat + (size_t)e * DD + lane * 4);
        float* p = Sacc + dbase;
        atomicAdd(p + 0, ev[0]); atomicAdd(p + 1, ev[1]);
        atomicAdd(p + 2, ev[2]); atomicAdd(p + 3, ev[3]);
    }
    if constexpr (HAS_X) {
        int s = src[e];
        const v4f xv = *(const v4f*)(Xfeat + (size_t)s * DD + lane * 4);
        float* p = Gacc + dbase;
        atomicAdd(p + 0, xv[0]); atomicAdd(p + 1, xv[1]);
        atomicAdd(p + 2, xv[2]); atomicAdd(p + 3, xv[3]);
    }
    if constexpr (HAS_CNT) {
        if (lane == 0) atomicAdd(cnt + d, 1.0f);
    }
}

// ---------------- per-graph node counts ------------------------------------
__global__ void node_counts(const int* __restrict__ gids, float* __restrict__ counts, int n) {
    int t = blockIdx.x * blockDim.x + threadIdx.x;
    if (t < n) atomicAdd(counts + gids[t], 1.0f);
}

// ---------------- fused WMMA GEMM ------------------------------------------
// Out = epilogue( (A0 [+ A1]) @ Wt^T  + rowScale(row)*bias[col] [+ Padd] )
// One wave per 16x16 C tile; block = 8 waves = one 16-row stripe, all N tiles.
// K = 128 -> 4x v_wmma_f32_16x16x32_bf16 (f32 accumulate).
// All feature flags are template params -> straight-line vectorized loads.
template <bool HAS_A1, bool HAS_RS, bool HAS_PADD, bool DO_SILU, bool DO_POOL>
__global__ void gemm_bf16(const float* __restrict__ A0, const float* __restrict__ A1,
                          const __bf16* __restrict__ Wt, const float* __restrict__ bias,
                          const float* __restrict__ rowScale, const float* __restrict__ Padd,
                          float* __restrict__ Out, const int* __restrict__ gids,
                          float* __restrict__ pooled) {
    const int m0   = blockIdx.x * 16;
    const int n0   = (threadIdx.x >> 5) * 16;     // wave id -> N tile
    const int lane = threadIdx.x & 31;
    const int half = lane >> 4;                   // lane-group (K-offset select)
    const int r    = lane & 15;                   // row (A) / col (B) in tile

    const float*  aRow  = A0 + (size_t)(m0 + r) * DD;
    const float*  a1Row = HAS_A1 ? (A1 + (size_t)(m0 + r) * DD) : nullptr;
    const __bf16* bRow  = Wt + (size_t)(n0 + r) * DD;

    v8f c = {};
#pragma unroll
    for (int kb = 0; kb < 4; ++kb) {
        const int ko = kb * 32 + half * 8;        // ISA 16-bit A layout: two
                                                  // contiguous 8-elem K chunks
        // ---- A fragment: 16 f32 in four b128 loads, optional residual add
        v4f p0 = *(const v4f*)(aRow + ko);
        v4f p1 = *(const v4f*)(aRow + ko + 4);
        v4f p2 = *(const v4f*)(aRow + ko + 16);
        v4f p3 = *(const v4f*)(aRow + ko + 20);
        if constexpr (HAS_A1) {
            p0 += *(const v4f*)(a1Row + ko);
            p1 += *(const v4f*)(a1Row + ko + 4);
            p2 += *(const v4f*)(a1Row + ko + 16);
            p3 += *(const v4f*)(a1Row + ko + 20);
        }
        v16bf a;
#pragma unroll
        for (int j = 0; j < 4; ++j) {
            a[j]      = (__bf16)p0[j];
            a[4 + j]  = (__bf16)p1[j];
            a[8 + j]  = (__bf16)p2[j];
            a[12 + j] = (__bf16)p3[j];
        }
        // ---- B fragment: 16 bf16 in two b128 loads (pre-transposed weights)
        v8bf b0 = *(const v8bf*)(bRow + ko);
        v8bf b1 = *(const v8bf*)(bRow + ko + 16);
        v16bf b = __builtin_shufflevector(b0, b1, 0, 1, 2, 3, 4, 5, 6, 7,
                                                  8, 9, 10, 11, 12, 13, 14, 15);
        c = __builtin_amdgcn_wmma_f32_16x16x32_bf16(
                false, a, false, b, (short)0, c, false, false);
    }

    // C/D layout: VGPR i, lane -> row = m0 + half*8 + i, col = n0 + (lane&15)
    const int col = n0 + r;
    const float bcol = bias[col];
#pragma unroll
    for (int i = 0; i < 8; ++i) {
        const int row = m0 + half * 8 + i;
        float v = c[i];
        if constexpr (HAS_RS) v += rowScale[row] * bcol;
        else                  v += bcol;
        if constexpr (HAS_PADD) v += Padd[(size_t)row * DD + col];
        if constexpr (DO_SILU) v = v * (1.0f / (1.0f + __expf(-v)));
        if constexpr (DO_POOL) atomicAdd(&pooled[(size_t)gids[row] * DD + col], v);
        else                   Out[(size_t)row * DD + col] = v;
    }
}

// ---------------- readout: out[g] = (pooled[g]/max(cnt,1)) . fc_w + fc_b ----
__global__ void readout(const float* __restrict__ pooled, const float* __restrict__ counts,
                        const float* __restrict__ fcw, const float* __restrict__ fcb,
                        float* __restrict__ out) {
    int g = blockIdx.x * (blockDim.x >> 5) + (threadIdx.x >> 5);
    if (g >= NGRAPH) return;
    int lane = threadIdx.x & 31;
    float acc = 0.0f;
#pragma unroll
    for (int j = 0; j < 4; ++j) {
        int idx = lane + 32 * j;
        acc += pooled[(size_t)g * DD + idx] * fcw[idx];
    }
#pragma unroll
    for (int off = 16; off > 0; off >>= 1) acc += __shfl_down(acc, off);
    if (lane == 0) {
        float cg = counts[g];
        if (cg < 1.0f) cg = 1.0f;
        out[g] = acc / cg + fcb[0];
    }
}

// ---------------------------------------------------------------------------
extern "C" void kernel_launch(void* const* d_in, const int* in_sizes, int n_in,
                              void* d_out, int out_size, void* d_ws, size_t ws_size,
                              hipStream_t stream) {
    (void)in_sizes; (void)n_in; (void)out_size; (void)ws_size;

    const float* x0  = (const float*)d_in[0];
    const float* e0  = (const float*)d_in[1];
    const int*   src = (const int*)d_in[2];
    const int*   dst = (const int*)d_in[3];
    const int*   gid = (const int*)d_in[4];
    const float* We1 = (const float*)d_in[5];  const float* be1 = (const float*)d_in[6];
    const float* Wf1 = (const float*)d_in[7];  const float* bf1 = (const float*)d_in[8];
    const float* We2 = (const float*)d_in[9];  const float* be2 = (const float*)d_in[10];
    const float* Wf2 = (const float*)d_in[11]; const float* bf2 = (const float*)d_in[12];
    const float* fcw = (const float*)d_in[13]; const float* fcb = (const float*)d_in[14];
    float* out = (float*)d_out;

    // workspace layout
    char* ws = (char*)d_ws;
    const size_t NDb = (size_t)NNODES * DD * sizeof(float);   // 25.6 MB
    float* S    = (float*)(ws + 0 * NDb);   // segsum(e0, dst)
    float* T1   = (float*)(ws + 1 * NDb);   // segsum(e1, dst) = S@We1 + deg*be1
    float* Gacc = (float*)(ws + 2 * NDb);   // segsum(x[src], dst)  (reused)
    float* X    = (float*)(ws + 3 * NDb);   // x1
    float* H2   = (float*)(ws + 4 * NDb);   // pre-activation layer 2
    size_t base = 5 * NDb;
    float* cnt    = (float*)(ws + base); base += (size_t)NNODES * sizeof(float);
    float* pooled = (float*)(ws + base); base += (size_t)NGRAPH * DD * sizeof(float);
    float* counts = (float*)(ws + base); base += (size_t)NGRAPH * sizeof(float);
    __bf16* Wt    = (__bf16*)(ws + base);   // 4 pre-transposed bf16 weights

    hipMemsetAsync(S,      0, NDb,                     stream);
    hipMemsetAsync(Gacc,   0, NDb,                     stream);
    hipMemsetAsync(cnt,    0, (size_t)NNODES * 4,      stream);
    hipMemsetAsync(pooled, 0, (size_t)NGRAPH * DD * 4, stream);
    hipMemsetAsync(counts, 0, (size_t)NGRAPH * 4,      stream);

    prep_weights<<<(4 * DD * DD + 255) / 256, 256, 0, stream>>>(We1, Wf1, We2, Wf2, Wt);

    // pass 1: S += e0[dst], G += x0[src], deg
    scatter_edges<true, true, true><<<(NEDGES + 7) / 8, 256, 0, stream>>>(
        e0, x0, src, dst, S, Gacc, cnt, NEDGES);
    node_counts<<<(NNODES + 255) / 256, 256, 0, stream>>>(gid, counts, NNODES);

    const int MT = NNODES / 16;  // 3125 exact
    // T1 = S@We1 + deg*be1        (== segsum of layer-1 edge features)
    gemm_bf16<false, true, false, false, false><<<MT, 256, 0, stream>>>(
        S, nullptr, Wt + 0 * DD * DD, be1, cnt, nullptr, T1, nullptr, nullptr);
    // x1 = silu((T1 + G)@Wf1 + bf1)
    gemm_bf16<true, false, false, true, false><<<MT, 256, 0, stream>>>(
        T1, Gacc, Wt + 1 * DD * DD, bf1, nullptr, nullptr, X, nullptr, nullptr);

    // pass 2: G = segsum(x1[src], dst)
    hipMemsetAsync(Gacc, 0, NDb, stream);
    scatter_edges<false, true, false><<<(NEDGES + 7) / 8, 256, 0, stream>>>(
        nullptr, X, src, dst, nullptr, Gacc, nullptr, NEDGES);

    // H2 = T1@We2 + deg*be2 + G
    gemm_bf16<false, true, true, false, false><<<MT, 256, 0, stream>>>(
        T1, nullptr, Wt + 2 * DD * DD, be2, cnt, Gacc, H2, nullptr, nullptr);
    // x2 = silu(H2@Wf2 + bf2), fused mean-pool accumulation into pooled[]
    gemm_bf16<false, false, false, true, true><<<MT, 256, 0, stream>>>(
        H2, nullptr, Wt + 3 * DD * DD, bf2, nullptr, nullptr, nullptr, gid, pooled);

    readout<<<NGRAPH / 8, 256, 0, stream>>>(pooled, counts, fcw, fcb, out);
}
